// MultiHeadSelfAttention_24790551232957
// MI455X (gfx1250) — compile-verified, matrix operand order checked by reading, not run on hardware
//
#include <hip/hip_runtime.h>
#include <hip/hip_bf16.h>
#include <math.h>

typedef __bf16 bf16;
typedef __attribute__((ext_vector_type(16))) __bf16 v16bf;
typedef __attribute__((ext_vector_type(8)))  __bf16 v8bf;
typedef __attribute__((ext_vector_type(8)))  float  v8f;
typedef __attribute__((ext_vector_type(4)))  float  v4f;

#define S_LEN  1024
#define DMODEL 1024
#define NHEADS 16
#define DK     64
#define BATCH  8
#define NTOK   (BATCH * S_LEN)   /* 8192 token rows */
#define LN_EPS 1e-5f

// ---------------------------------------------------------------------------
// WMMA helpers.  Per-lane fragment pattern for 16-bit A/B (ISA 7.12.2):
//   lanes 0-15 : row = lane,      K = base+0..7  and base+16..23
//   lanes 16-31: row = lane-16,   K = base+8..15 and base+24..31
// i.e. two contiguous 16-byte chunks at p and p+16 (p already includes the
// +0/+8 half-wave offset).
// ---------------------------------------------------------------------------
__device__ __forceinline__ v16bf load_frag16(const bf16* p) {
  v8bf a = *(const v8bf*)(p);
  v8bf b = *(const v8bf*)(p + 16);
  v16bf r;
#pragma unroll
  for (int i = 0; i < 8; ++i) { r[i] = a[i]; r[i + 8] = b[i]; }
  return r;
}

__device__ __forceinline__ v8f wmma_bf16(v16bf a, v16bf b, v8f c) {
  return __builtin_amdgcn_wmma_f32_16x16x32_bf16(false, a, false, b,
                                                 (short)0, c, false, false);
}

// ---------------------------------------------------------------------------
// Kernel 0: fp32 -> bf16 cast
// ---------------------------------------------------------------------------
__global__ void cast_f32_bf16(const float* __restrict__ in,
                              bf16* __restrict__ out, int n) {
  int i = blockIdx.x * blockDim.x + threadIdx.x;
  if (i < n) out[i] = (bf16)in[i];
}

// ---------------------------------------------------------------------------
// Kernel 1: QKV projection  out = x @ W^T + b   (M=8192, N=1024, K=1024)
// Block = 128 thr (4 waves); wave computes 16x64; block tile 16x256.
// grid = (1024/256, 8192/16)
// mode 0: write [b,h,s,d]   (Q and K layout)
// mode 1: write [b,h,d,s]   (V transposed, so PV B-rows are contiguous)
// ---------------------------------------------------------------------------
__global__ void qkv_gemm(const bf16* __restrict__ Xb,
                         const bf16* __restrict__ Wb,
                         const float* __restrict__ bias,
                         bf16* __restrict__ dst, int mode) {
  const int lane = threadIdx.x & 31;
  const int wave = threadIdx.x >> 5;
  const int lr   = lane & 15;
  const int hi   = (lane >> 4) ? 8 : 0;
  const int row0 = blockIdx.y * 16;
  const int col0 = blockIdx.x * 256 + wave * 64;

  v8f acc[4] = {v8f{}, v8f{}, v8f{}, v8f{}};
  for (int kk = 0; kk < DMODEL / 32; ++kk) {
    const int kb = kk * 32 + hi;
    v16bf a = load_frag16(Xb + (size_t)(row0 + lr) * DMODEL + kb);
#pragma unroll
    for (int t = 0; t < 4; ++t) {
      v16bf b = load_frag16(Wb + (size_t)(col0 + t * 16 + lr) * DMODEL + kb);
      acc[t] = wmma_bf16(a, b, acc[t]);
    }
  }
  // epilogue: C vgpr j -> row = j + (lane>=16 ? 8 : 0), col = lane%16
  const int rsub = (lane >> 4) ? 8 : 0;
#pragma unroll
  for (int t = 0; t < 4; ++t) {
    const int col = col0 + t * 16 + lr;
    const float bv = bias[col];
    const int h = col >> 6, d = col & 63;
#pragma unroll
    for (int j = 0; j < 8; ++j) {
      const int row = row0 + rsub + j;
      const int bb = row >> 10, s = row & 1023;
      const float v = acc[t][j] + bv;
      size_t off;
      if (mode == 0)
        off = ((size_t)(bb * NHEADS + h) * S_LEN + s) * DK + d;
      else
        off = ((size_t)(bb * NHEADS + h) * DK + d) * S_LEN + s;
      dst[off] = (bf16)v;
    }
  }
}

// ---------------------------------------------------------------------------
// Kernel 2: fused attention for one (b,h, 16-query) block.
// Block = 128 thr (4 waves), grid = B*H*64 = 8192.
// Dyn LDS: sc[16*1024] f32 scores + red[128] + stat[16] + pb[16*1024] bf16.
// Phase 1: scores = Q K^T * 1/8 via WMMA -> LDS (f32)
// Phase 2: block softmax; write fp32 attn to d_out with b128 stores (the
//          roofline-dominant 536MB write) and stash bf16 P in LDS.
// Phase 3: ctx = P V via WMMA; A-frags come from bf16 LDS as 2x ds_load_b128.
// ---------------------------------------------------------------------------
__global__ void attn_fused(const bf16* __restrict__ Qb,
                           const bf16* __restrict__ Kb,
                           const bf16* __restrict__ Vtb,
                           bf16* __restrict__ ctxb,
                           float* __restrict__ attnOut) {
  extern __shared__ float smem[];
  float* sc   = smem;                    // 16*1024 f32
  float* red  = smem + 16 * 1024;        // 128
  float* stat = red + 128;               // 16
  bf16*  pb   = (bf16*)(stat + 16);      // 16*1024 bf16 (16B aligned)

  const int lane = threadIdx.x & 31;
  const int wave = threadIdx.x >> 5;
  const int lr   = lane & 15;
  const int hi   = (lane >> 4) ? 8 : 0;
  const int rsub = (lane >> 4) ? 8 : 0;

  const int bid = blockIdx.x;
  const int qt  = bid & 63;   // query tile in sequence
  const int bh  = bid >> 6;   // b*16+h
  const int q0  = qt * 16;

  // ---- Phase 1: scores ----
  const bf16* Qp = Qb + ((size_t)bh * S_LEN + q0) * DK;
  const bf16* Kp = Kb + (size_t)bh * S_LEN * DK;
  v16bf aq0 = load_frag16(Qp + (size_t)lr * DK + 0 + hi);
  v16bf aq1 = load_frag16(Qp + (size_t)lr * DK + 32 + hi);
#pragma unroll 1
  for (int t = 0; t < 16; ++t) {
    const int ct = wave * 16 + t;           // column tile (key tile)
    const bf16* kp = Kp + (size_t)(ct * 16 + lr) * DK;
    v8f c = {};
    c = wmma_bf16(aq0, load_frag16(kp + hi), c);
    c = wmma_bf16(aq1, load_frag16(kp + 32 + hi), c);
#pragma unroll
    for (int j = 0; j < 8; ++j)
      sc[(rsub + j) * 1024 + ct * 16 + lr] = c[j] * 0.125f;  // 1/sqrt(64)
  }
  __syncthreads();

  // ---- Phase 2: softmax over 1024 keys per row ----
  const int r   = threadIdx.x >> 3;   // 0..15
  const int seg = threadIdx.x & 7;    // 128 cols per (row,seg)
  float m = -INFINITY;
  for (int c = seg * 128; c < seg * 128 + 128; ++c)
    m = fmaxf(m, sc[r * 1024 + c]);
  red[r * 8 + seg] = m;
  __syncthreads();
  if (threadIdx.x < 16) {
    float mm = red[threadIdx.x * 8];
    for (int i = 1; i < 8; ++i) mm = fmaxf(mm, red[threadIdx.x * 8 + i]);
    stat[threadIdx.x] = mm;
  }
  __syncthreads();
  const float rm = stat[r];
  float ssum = 0.f;
  for (int c = seg * 128; c < seg * 128 + 128; ++c) {
    float e = __expf(sc[r * 1024 + c] - rm);
    sc[r * 1024 + c] = e;
    ssum += e;
  }
  __syncthreads();   // all reads of stat done
  red[r * 8 + seg] = ssum;
  __syncthreads();
  if (threadIdx.x < 16) {
    float s = 0.f;
    for (int i = 0; i < 8; ++i) s += red[threadIdx.x * 8 + i];
    stat[threadIdx.x] = 1.0f / s;
  }
  __syncthreads();

  // normalize: vectorized fp32 attn write (b128) + bf16 P into LDS.
  // thread t owns 8 contiguous cols [t*8, t*8+8).
  float* aout = attnOut + ((size_t)bh * S_LEN + q0) * S_LEN;
  const int c0 = threadIdx.x * 8;
#pragma unroll 1
  for (int rr = 0; rr < 16; ++rr) {
    const float inv = stat[rr];
    float v[8];
#pragma unroll
    for (int i = 0; i < 8; ++i) v[i] = sc[rr * 1024 + c0 + i] * inv;
    v4f s0 = {v[0], v[1], v[2], v[3]};
    v4f s1 = {v[4], v[5], v[6], v[7]};
    *(v4f*)(aout + (size_t)rr * S_LEN + c0)     = s0;
    *(v4f*)(aout + (size_t)rr * S_LEN + c0 + 4) = s1;
    v8bf pv;
#pragma unroll
    for (int i = 0; i < 8; ++i) pv[i] = (bf16)v[i];
    *(v8bf*)(pb + rr * 1024 + c0) = pv;
  }
  __syncthreads();

  // ---- Phase 3: ctx = P @ V  (one 16x16 d-tile per wave) ----
  const int dcol = wave * 16 + lr;                 // V^T row = head dim
  const bf16* vp = Vtb + ((size_t)bh * DK + dcol) * S_LEN;
  v8f acc = {};
#pragma unroll 1
  for (int kk = 0; kk < S_LEN / 32; ++kk) {
    const int kb = kk * 32 + hi;
    v16bf a = load_frag16(pb + lr * 1024 + kb);    // 2x ds_load_b128
    v16bf b = load_frag16(vp + kb);
    acc = wmma_bf16(a, b, acc);
  }
  const int bb = bh >> 4, h = bh & 15;
#pragma unroll
  for (int j = 0; j < 8; ++j) {
    const int s_tok = q0 + rsub + j;
    ctxb[((size_t)(bb * S_LEN + s_tok)) * DMODEL + h * DK + dcol] = (bf16)acc[j];
  }
}

// ---------------------------------------------------------------------------
// Kernel 3: out = ctx @ Wo^T + bo ; h = out + x ; y = LayerNorm(h)
// Block = 256 thr (8 waves), grid = 8192/16 = 512 row blocks.
// Each wave computes 8 col tiles (16x128); strip h[16x1024] kept in LDS.
// ---------------------------------------------------------------------------
__global__ void oproj_ln(const bf16* __restrict__ Cb,
                         const bf16* __restrict__ Wob,
                         const float* __restrict__ bo,
                         const float* __restrict__ x,
                         const float* __restrict__ gamma,
                         const float* __restrict__ beta,
                         float* __restrict__ y) {
  extern __shared__ float smem[];
  float* hbuf = smem;               // 16*1024
  float* red  = smem + 16 * 1024;   // 256 (sum)
  float* redq = red + 256;          // 256 (sumsq)
  float* muA  = redq + 256;         // 16
  float* rsA  = muA + 16;           // 16

  const int lane = threadIdx.x & 31;
  const int wave = threadIdx.x >> 5;
  const int lr   = lane & 15;
  const int hi   = (lane >> 4) ? 8 : 0;
  const int rsub = (lane >> 4) ? 8 : 0;
  const int row0 = blockIdx.x * 16;

  v8f acc[8] = {v8f{}, v8f{}, v8f{}, v8f{}, v8f{}, v8f{}, v8f{}, v8f{}};
#pragma unroll 1
  for (int kk = 0; kk < DMODEL / 32; ++kk) {
    const int kb = kk * 32 + hi;
    v16bf a = load_frag16(Cb + (size_t)(row0 + lr) * DMODEL + kb);
#pragma unroll
    for (int t = 0; t < 8; ++t) {
      v16bf b = load_frag16(Wob + (size_t)(wave * 128 + t * 16 + lr) * DMODEL + kb);
      acc[t] = wmma_bf16(a, b, acc[t]);
    }
  }
#pragma unroll
  for (int t = 0; t < 8; ++t) {
    const int col = wave * 128 + t * 16 + lr;
    const float bv = bo[col];
#pragma unroll
    for (int j = 0; j < 8; ++j) {
      const int row = row0 + rsub + j;
      hbuf[(rsub + j) * 1024 + col] =
          acc[t][j] + bv + x[(size_t)row * DMODEL + col];
    }
  }
  __syncthreads();

  // LayerNorm: 16 threads per row, 64 cols per thread
  const int r  = threadIdx.x >> 4;
  const int sg = threadIdx.x & 15;
  float sum = 0.f, sq = 0.f;
  for (int c = sg * 64; c < sg * 64 + 64; ++c) {
    float v = hbuf[r * 1024 + c];
    sum += v; sq += v * v;
  }
  red[r * 16 + sg]  = sum;
  redq[r * 16 + sg] = sq;
  __syncthreads();
  if (threadIdx.x < 16) {
    float s = 0.f, q = 0.f;
    for (int i = 0; i < 16; ++i) { s += red[threadIdx.x * 16 + i]; q += redq[threadIdx.x * 16 + i]; }
    float mu  = s * (1.0f / DMODEL);
    float var = q * (1.0f / DMODEL) - mu * mu;
    muA[threadIdx.x] = mu;
    rsA[threadIdx.x] = rsqrtf(var + LN_EPS);
  }
  __syncthreads();
  // vectorized y write: thread t owns cols [t*4, t*4+4)
  const int c0 = threadIdx.x * 4;
#pragma unroll 1
  for (int rr = 0; rr < 16; ++rr) {
    const float mu = muA[rr], rs = rsA[rr];
    v4f o;
#pragma unroll
    for (int i = 0; i < 4; ++i)
      o[i] = (hbuf[rr * 1024 + c0 + i] - mu) * rs * gamma[c0 + i] + beta[c0 + i];
    *(v4f*)(y + (size_t)(row0 + rr) * DMODEL + c0) = o;
  }
}

// ---------------------------------------------------------------------------
extern "C" void kernel_launch(void* const* d_in, const int* in_sizes, int n_in,
                              void* d_out, int out_size, void* d_ws, size_t ws_size,
                              hipStream_t stream) {
  const float* x     = (const float*)d_in[0];
  const float* Wq    = (const float*)d_in[1];
  const float* bq    = (const float*)d_in[2];
  const float* Wk    = (const float*)d_in[3];
  const float* bk    = (const float*)d_in[4];
  const float* Wv    = (const float*)d_in[5];
  const float* bv    = (const float*)d_in[6];
  const float* Wo    = (const float*)d_in[7];
  const float* bo    = (const float*)d_in[8];
  const float* gamma = (const float*)d_in[9];
  const float* beta  = (const float*)d_in[10];

  float* y    = (float*)d_out;                                  // [8,1024,1024]
  float* attn = y + (size_t)BATCH * S_LEN * DMODEL;             // [8,16,1024,1024]

  // workspace partition (bf16 elements)
  bf16* p   = (bf16*)d_ws;
  bf16* xb  = p; p += (size_t)NTOK * DMODEL;
  bf16* wqb = p; p += (size_t)DMODEL * DMODEL;
  bf16* wkb = p; p += (size_t)DMODEL * DMODEL;
  bf16* wvb = p; p += (size_t)DMODEL * DMODEL;
  bf16* wob = p; p += (size_t)DMODEL * DMODEL;
  bf16* qb  = p; p += (size_t)NTOK * DMODEL;   // [b,h,s,d]
  bf16* kb  = p; p += (size_t)NTOK * DMODEL;   // [b,h,s,d]
  bf16* vtb = p; p += (size_t)NTOK * DMODEL;   // [b,h,d,s]
  bf16* cb  = p; p += (size_t)NTOK * DMODEL;   // ctx [b,s,D]

  // casts to bf16
  {
    const int nx = NTOK * DMODEL;
    cast_f32_bf16<<<(nx + 255) / 256, 256, 0, stream>>>(x, xb, nx);
    const int nw = DMODEL * DMODEL;
    cast_f32_bf16<<<(nw + 255) / 256, 256, 0, stream>>>(Wq, wqb, nw);
    cast_f32_bf16<<<(nw + 255) / 256, 256, 0, stream>>>(Wk, wkb, nw);
    cast_f32_bf16<<<(nw + 255) / 256, 256, 0, stream>>>(Wv, wvb, nw);
    cast_f32_bf16<<<(nw + 255) / 256, 256, 0, stream>>>(Wo, wob, nw);
  }

  // QKV projections
  {
    dim3 grid(DMODEL / 256, NTOK / 16);
    qkv_gemm<<<grid, 128, 0, stream>>>(xb, wqb, bq, qb, 0);
    qkv_gemm<<<grid, 128, 0, stream>>>(xb, wkb, bk, kb, 0);
    qkv_gemm<<<grid, 128, 0, stream>>>(xb, wvb, bv, vtb, 1);
  }

  // fused attention
  {
    const size_t shbytes =
        (size_t)(16 * 1024 + 128 + 16) * sizeof(float) +
        (size_t)(16 * 1024) * sizeof(bf16);
    attn_fused<<<BATCH * NHEADS * (S_LEN / 16), 128, shbytes, stream>>>(
        qb, kb, vtb, cb, attn);
  }

  // O projection + residual + LayerNorm
  {
    const size_t shbytes = (size_t)(16 * 1024 + 256 + 256 + 16 + 16) * sizeof(float);
    oproj_ln<<<NTOK / 16, 256, shbytes, stream>>>(cb, wob, bo, x, gamma, beta, y);
  }
}
